// TransformerBlock_25494925869431
// MI455X (gfx1250) — matrix operand order repaired and checked by reading, not compiled, not run on hardware
//
#include <hip/hip_runtime.h>
#include <hip/hip_bf16.h>

// ---------------------------------------------------------------------------
// Fixed problem dims (reference: EMBED=2048, HEADS=16, DFF=8192, B=2, N=2048)
// ---------------------------------------------------------------------------
#define BATCH 2
#define NTOK  2048
#define EMBED 2048
#define HEADS 16
#define HDIM  128
#define DFF   8192
#define MROWS (BATCH * NTOK)   // 4096

typedef _Float16 v16h __attribute__((ext_vector_type(16)));
typedef _Float16 v8h  __attribute__((ext_vector_type(8)));
typedef _Float16 v4h  __attribute__((ext_vector_type(4)));
typedef float    v8f  __attribute__((ext_vector_type(8)));
typedef unsigned int u32x4 __attribute__((ext_vector_type(4)));
typedef int          i32x8 __attribute__((ext_vector_type(8)));
typedef int          i32x4 __attribute__((ext_vector_type(4)));

#if __has_builtin(__builtin_amdgcn_tensor_load_to_lds) && \
    __has_builtin(__builtin_amdgcn_s_wait_tensorcnt)
#define HAVE_TDM 1
#else
#define HAVE_TDM 0
#endif

__device__ __forceinline__ v16h pack16(v8h a, v8h b) {
    v16h r;
#pragma unroll
    for (int i = 0; i < 8; ++i) { r[i] = a[i]; r[i + 8] = b[i]; }
    return r;
}

__device__ __forceinline__ v8f wmma_f16(v16h a, v16h b, v8f c) {
    // D = A(16x32 f16) * B(32x16 f16) + C(16x16 f32)
    return __builtin_amdgcn_wmma_f32_16x16x32_f16(false, a, false, b,
                                                  (short)0, c, false, false);
}

__device__ __forceinline__ float gelu_exact(float x) {
    return 0.5f * x * (1.0f + erff(x * 0.70710678118654752f));
}

#if HAVE_TDM
// ---------------------------------------------------------------------------
// TDM: DMA a 2D f16 tile [rows=64][cols=32] (row stride = K elems) into LDS.
// D# per CDNA5 ISA ch.8: group0 = {count/type, lds_addr, global_addr},
// group1 = {data_size, tensor dims, tile dims, dim0 stride}.
// Tile is always fully in-bounds -> tensor dims just need to cover the tile.
// This toolchain exposes the 6-arg builtin:
//   (uint32x4 g0, int32x8 g1, int32x4 g2, int32x4 g3, int32x8 gx, i32 cpol)
// ---------------------------------------------------------------------------
__device__ __forceinline__ void tdm_load_b_tile(const _Float16* gptr,
                                                unsigned lds_off, int K) {
    unsigned long long ga = (unsigned long long)(uintptr_t)gptr;
    u32x4 g0;
    g0[0] = 1u;                                       // count=1, user descriptor
    g0[1] = lds_off;                                  // LDS byte address
    g0[2] = (unsigned)ga;                             // global addr [31:0]
    g0[3] = (unsigned)((ga >> 32) & 0x01FFFFFFu)      // global addr [56:32]
            | (2u << 30);                             // type = 2 ("image")
    i32x8 g1;
    g1[0] = 1 << 16;                                  // data_size = 2 bytes
    g1[1] = (K & 0xFFFF) << 16;                       // tensor_dim0[15:0]
    g1[2] = ((K >> 16) & 0xFFFF) | (0xFFFF << 16);    // tensor_dim0[31:16] | tensor_dim1 lo
    g1[3] = (32 << 16);                               // tensor_dim1 hi=0 | tile_dim0=32
    g1[4] = 64;                                       // tile_dim1=64 | tile_dim2=0
    g1[5] = K;                                        // tensor_dim0_stride[31:0]
    g1[6] = 0;                                        // stride hi | dim1_stride lo
    g1[7] = 0;
    i32x4 gz4 = {0, 0, 0, 0};
    i32x8 gz8 = {0, 0, 0, 0, 0, 0, 0, 0};
    __builtin_amdgcn_tensor_load_to_lds(g0, g1, gz4, gz4, gz8, 0);
}
#endif

// ---------------------------------------------------------------------------
// f32 -> f16 elementwise convert (4 elems/thread, vectorized)
// ---------------------------------------------------------------------------
__global__ __launch_bounds__(256) void f32_to_f16_kernel(
    const float* __restrict__ in, _Float16* __restrict__ out, size_t n) {
    size_t i = ((size_t)blockIdx.x * 256 + threadIdx.x) * 4;
    if (i + 3 < n) {
        float4 f = *(const float4*)(in + i);
        v4h o;
        o[0] = (_Float16)f.x; o[1] = (_Float16)f.y;
        o[2] = (_Float16)f.z; o[3] = (_Float16)f.w;
        *(v4h*)(out + i) = o;
    }
}

// ---------------------------------------------------------------------------
// Tiled transpose + convert: W [K x N] f32 -> WT [N x K] f16
// ---------------------------------------------------------------------------
__global__ __launch_bounds__(256) void transpose_f16_kernel(
    const float* __restrict__ W, _Float16* __restrict__ WT, int K, int N) {
    __shared__ float tile[32][33];
    const int tx = threadIdx.x, ty = threadIdx.y;       // 32 x 8
    const int n0 = blockIdx.x * 32, k0 = blockIdx.y * 32;
#pragma unroll
    for (int i = 0; i < 32; i += 8)
        tile[ty + i][tx] = W[(size_t)(k0 + ty + i) * N + (n0 + tx)];
    __syncthreads();
#pragma unroll
    for (int i = 0; i < 32; i += 8)
        WT[(size_t)(n0 + ty + i) * K + (k0 + tx)] = (_Float16)tile[tx][ty + i];
}

// ---------------------------------------------------------------------------
// WMMA GEMM: out[M,N] = A[M,K](f16) * Bt[N,K]^T(f16) + bias[N]
// Block: 128 threads (4 waves). Block tile 128x64; each wave 32 rows x 64 cols
// (2 A-tiles x 4 B-tiles = 8 accumulators). K-loop step 32.
// B panel (64 x 32 f16 = 4KB) staged once per block into LDS, double-buffered
// via the Tensor Data Mover (wave 0 issues tensor_load_to_lds, waits on
// TENSORcnt, workgroup barrier publishes). Fallback: cooperative VMEM->LDS.
// MODE 0: f32 store          MODE 1: f16 store
// MODE 2: f16 store in attention-V layout [b,h,hd,NTOK]
// MODE 3: GELU then f16 store
// ---------------------------------------------------------------------------
template <int MODE>
__global__ __launch_bounds__(128) void gemm_f16_kernel(
    const _Float16* __restrict__ A, const _Float16* __restrict__ Bt,
    const float* __restrict__ bias, void* __restrict__ outp,
    int M, int N, int K) {
    __shared__ __align__(16) _Float16 lds_b[2][64][32];
    const int wave = threadIdx.x >> 5;
    const int lane = threadIdx.x & 31;
    const int lo = lane & 15, hi = lane >> 4;
    const int tM = blockIdx.x * 128 + wave * 32;
    const int tN = blockIdx.y * 64;
    const int nk = K / 32;

    v8f acc[2][4] = {};

    // A layout: lane row = lo; per-lane K = hi*8 + {0..7} and hi*8+16 + {0..7}
    const _Float16* arow0 = A + (size_t)(tM + lo) * K + hi * 8;
    const _Float16* arow1 = arow0 + (size_t)16 * K;
    const _Float16* bpanel = Bt + (size_t)tN * K;
    // Low 32 bits of a generic LDS pointer == workgroup-relative LDS offset.
    const unsigned ldsb = (unsigned)(uintptr_t)(void*)&lds_b[0][0][0];
    (void)ldsb;

#if HAVE_TDM
    if (wave == 0) tdm_load_b_tile(bpanel, ldsb, K);
#endif
    for (int i = 0; i < nk; ++i) {
        const int k = i * 32;
#if HAVE_TDM
        const int cur = i & 1;
        if (wave == 0) {
            if (i + 1 < nk) {
                tdm_load_b_tile(bpanel + (k + 32), ldsb + (cur ^ 1) * 4096, K);
                __builtin_amdgcn_s_wait_tensorcnt(1);   // previous tile done
            } else {
                __builtin_amdgcn_s_wait_tensorcnt(0);
            }
        }
#else
        const int cur = 0;
        {   // cooperative stage: 4KB = 256 x 16B chunks, 2 per thread
#pragma unroll
            for (int j = 0; j < 2; ++j) {
                const int c = threadIdx.x + j * 128;
                const int row = c >> 2, off = (c & 3) * 8;
                *(v8h*)&lds_b[0][row][off] =
                    *(const v8h*)(bpanel + (size_t)row * K + k + off);
            }
        }
#endif
        __syncthreads();
        v16h a0 = pack16(*(const v8h*)(arow0 + k), *(const v8h*)(arow0 + k + 16));
        v16h a1 = pack16(*(const v8h*)(arow1 + k), *(const v8h*)(arow1 + k + 16));
#pragma unroll
        for (int t = 0; t < 4; ++t) {
            // B layout: lane col = lo; per-lane K = hi*8 + {0..7} and hi*8+16 + {0..7}
            const _Float16* bls = &lds_b[cur][t * 16 + lo][hi * 8];
            v16h bv = pack16(*(const v8h*)bls, *(const v8h*)(bls + 16));
            acc[0][t] = wmma_f16(a0, bv, acc[0][t]);
            acc[1][t] = wmma_f16(a1, bv, acc[1][t]);
        }
        __syncthreads();
    }

    // C layout: VGPR r -> row (a*16 + r + 8*hi), col t*16 + lo
#pragma unroll
    for (int a = 0; a < 2; ++a) {
#pragma unroll
        for (int t = 0; t < 4; ++t) {
            const int col = tN + t * 16 + lo;
            const float bv = bias[col];
#pragma unroll
            for (int r = 0; r < 8; ++r) {
                const int row = tM + a * 16 + r + 8 * hi;
                float v = acc[a][t][r] + bv;
                if (MODE == 3) v = gelu_exact(v);
                if (MODE == 0) {
                    ((float*)outp)[(size_t)row * N + col] = v;
                } else if (MODE == 1 || MODE == 3) {
                    ((_Float16*)outp)[(size_t)row * N + col] = (_Float16)v;
                } else { // MODE 2: V^T for attention: [b, h, d, tok]
                    const int tok = row & (NTOK - 1);
                    const int bb  = row >> 11;
                    const int hh  = col >> 7;
                    const int dd  = col & (HDIM - 1);
                    ((_Float16*)outp)[(((size_t)(bb * HEADS + hh) * HDIM + dd) << 11) + tok] =
                        (_Float16)v;
                }
            }
        }
    }
}

// ---------------------------------------------------------------------------
// Flash attention: one wave = one 16-row q tile for one (b,h).
// Scores via Q(16x128)*K^T in 4 hd-chunks of 32; online softmax with
// shfl_xor(width=16) row reductions (C-layout row = r + 8*hi, col = lane&15);
// P re-layout C->A via per-wave LDS; P*V via 8 WMMAs over hd.
// ---------------------------------------------------------------------------
__global__ __launch_bounds__(128) void attn_kernel(
    const _Float16* __restrict__ Qb,  // [MROWS, EMBED]
    const _Float16* __restrict__ Kb,  // [MROWS, EMBED]
    const _Float16* __restrict__ VT,  // [B*H, HDIM, NTOK]
    _Float16* __restrict__ O) {       // [MROWS, EMBED]
    constexpr float kScale = 0.08838834764831845f;  // 1/sqrt(128)
    __shared__ __align__(16) _Float16 pls[4][16][32];

    const int wave = threadIdx.x >> 5;
    const int lane = threadIdx.x & 31;
    const int lo = lane & 15, hi = lane >> 4;
    const int bh = blockIdx.x;
    const int b = bh >> 4, h = bh & 15;
    const int q0 = blockIdx.y * 64 + wave * 16;

    // Preload Q in A-layout for all 4 hd-chunks
    const _Float16* qrow = Qb + (size_t)(b * NTOK + q0 + lo) * EMBED + h * HDIM + hi * 8;
    v16h qa[4];
#pragma unroll
    for (int kd = 0; kd < 4; ++kd)
        qa[kd] = pack16(*(const v8h*)(qrow + kd * 32),
                        *(const v8h*)(qrow + kd * 32 + 16));

    v8f accd[8] = {};
    float mrow[8], lrow[8];
#pragma unroll
    for (int r = 0; r < 8; ++r) { mrow[r] = -3.0e38f; lrow[r] = 0.0f; }

    const _Float16* kbase = Kb + (size_t)(b * NTOK) * EMBED + h * HDIM;
    const _Float16* vbase = VT + (size_t)bh * HDIM * NTOK;

    for (int kv = 0; kv < NTOK; kv += 32) {
        // ---- scores: two 16x16 tiles (kv..kv+15, kv+16..kv+31) ----
        v8f s0 = {}, s1 = {};
#pragma unroll
        for (int kd = 0; kd < 4; ++kd) {
            const _Float16* kr0 = kbase + (size_t)(kv + lo) * EMBED + kd * 32 + hi * 8;
            v16h b0 = pack16(*(const v8h*)kr0, *(const v8h*)(kr0 + 16));
            s0 = wmma_f16(qa[kd], b0, s0);
            const _Float16* kr1 = kr0 + (size_t)16 * EMBED;
            v16h b1 = pack16(*(const v8h*)kr1, *(const v8h*)(kr1 + 16));
            s1 = wmma_f16(qa[kd], b1, s1);
        }
        // ---- online softmax update (per row = r + 8*hi) ----
#pragma unroll
        for (int r = 0; r < 8; ++r) {
            float a0 = s0[r] * kScale;
            float a1 = s1[r] * kScale;
            float mx = fmaxf(a0, a1);
            mx = fmaxf(mx, __shfl_xor(mx, 1, 16));
            mx = fmaxf(mx, __shfl_xor(mx, 2, 16));
            mx = fmaxf(mx, __shfl_xor(mx, 4, 16));
            mx = fmaxf(mx, __shfl_xor(mx, 8, 16));
            const float mnew = fmaxf(mrow[r], mx);
            const float corr = __expf(mrow[r] - mnew);
            mrow[r] = mnew;
            const float p0 = __expf(a0 - mnew);
            const float p1 = __expf(a1 - mnew);
            float ps = p0 + p1;
            ps += __shfl_xor(ps, 1, 16);
            ps += __shfl_xor(ps, 2, 16);
            ps += __shfl_xor(ps, 4, 16);
            ps += __shfl_xor(ps, 8, 16);
            lrow[r] = lrow[r] * corr + ps;
#pragma unroll
            for (int d = 0; d < 8; ++d) accd[d][r] *= corr;
            const int row = r + 8 * hi;
            pls[wave][row][lo]      = (_Float16)p0;
            pls[wave][row][lo + 16] = (_Float16)p1;
        }
        // LDS is in-order per wave; make waits explicit & stop reordering
        asm volatile("s_wait_dscnt 0" ::: "memory");
        // re-read P in A-layout: lane row = lo, kv elems = hi*8 + {0..7,16..23}
        v16h pa = pack16(*(const v8h*)&pls[wave][lo][hi * 8],
                         *(const v8h*)&pls[wave][lo][hi * 8 + 16]);
        // ---- P * V over 8 dim-tiles ----
#pragma unroll
        for (int d = 0; d < 8; ++d) {
            const _Float16* vr = vbase + (size_t)(d * 16 + lo) * NTOK + kv + hi * 8;
            v16h vb = pack16(*(const v8h*)vr, *(const v8h*)(vr + 16));
            accd[d] = wmma_f16(pa, vb, accd[d]);
        }
        asm volatile("s_wait_dscnt 0" ::: "memory");
    }

    // ---- normalize and store (f16, row-major [tok, C]) ----
#pragma unroll
    for (int r = 0; r < 8; ++r) {
        const float inv = 1.0f / lrow[r];
        const int qr = q0 + r + 8 * hi;
        const size_t base = (size_t)(b * NTOK + qr) * EMBED + h * HDIM;
#pragma unroll
        for (int d = 0; d < 8; ++d)
            O[base + d * 16 + lo] = (_Float16)(accd[d][r] * inv);
    }
}

// ---------------------------------------------------------------------------
// Fused residual add + LayerNorm: out = LN(X + Y)*g + b.
// One block per row (2048 cols, 256 threads, 8 cols/thread).
// ---------------------------------------------------------------------------
__global__ __launch_bounds__(256) void add_ln_kernel(
    const float* __restrict__ X, const float* __restrict__ Y,
    const float* __restrict__ g, const float* __restrict__ bta,
    float* __restrict__ outf, _Float16* __restrict__ outh) {
    __shared__ float red[256];
    const int row = blockIdx.x, tid = threadIdx.x;
    const size_t base = (size_t)row * EMBED;
    float v[8];
    float sum = 0.0f, sq = 0.0f;
#pragma unroll
    for (int i = 0; i < 8; ++i) {
        const int c = i * 256 + tid;
        const float t = X[base + c] + Y[base + c];
        v[i] = t; sum += t; sq += t * t;
    }
    red[tid] = sum; __syncthreads();
    for (int s = 128; s > 0; s >>= 1) {
        if (tid < s) red[tid] += red[tid + s];
        __syncthreads();
    }
    sum = red[0]; __syncthreads();
    red[tid] = sq; __syncthreads();
    for (int s = 128; s > 0; s >>= 1) {
        if (tid < s) red[tid] += red[tid + s];
        __syncthreads();
    }
    sq = red[0];
    const float mean = sum * (1.0f / EMBED);
    const float var  = sq * (1.0f / EMBED) - mean * mean;
    const float rstd = rsqrtf(var + 1e-5f);
#pragma unroll
    for (int i = 0; i < 8; ++i) {
        const int c = i * 256 + tid;
        const float o = (v[i] - mean) * rstd * g[c] + bta[c];
        outf[base + c] = o;
        if (outh) outh[base + c] = (_Float16)o;
    }
}

// ---------------------------------------------------------------------------
// Orchestration
// ---------------------------------------------------------------------------
extern "C" void kernel_launch(void* const* d_in, const int* in_sizes, int n_in,
                              void* d_out, int out_size, void* d_ws, size_t ws_size,
                              hipStream_t stream) {
    (void)in_sizes; (void)n_in; (void)out_size; (void)ws_size;
    const float* x   = (const float*)d_in[0];
    const float* Wq  = (const float*)d_in[1];
    const float* bq  = (const float*)d_in[2];
    const float* Wk  = (const float*)d_in[3];
    const float* bk  = (const float*)d_in[4];
    const float* Wv  = (const float*)d_in[5];
    const float* bv  = (const float*)d_in[6];
    const float* Wo  = (const float*)d_in[7];
    const float* bo  = (const float*)d_in[8];
    const float* g1  = (const float*)d_in[9];
    const float* b1  = (const float*)d_in[10];
    const float* g2  = (const float*)d_in[11];
    const float* b2  = (const float*)d_in[12];
    const float* W1  = (const float*)d_in[13];
    const float* bf1 = (const float*)d_in[14];
    const float* W2  = (const float*)d_in[15];
    const float* bf2 = (const float*)d_in[16];

    constexpr size_t M  = MROWS;            // 4096
    constexpr size_t C  = EMBED;            // 2048
    constexpr size_t F  = DFF;              // 8192

    char* ws = (char*)d_ws;
    size_t cur = 0;
    auto alloc = [&](size_t bytes) -> char* {
        char* p = ws + cur;
        cur += (bytes + 255) & ~(size_t)255;
        return p;
    };
    _Float16* Xb  = (_Float16*)alloc(M * C * 2);
    _Float16* WqT = (_Float16*)alloc(C * C * 2);
    _Float16* WkT = (_Float16*)alloc(C * C * 2);
    _Float16* WvT = (_Float16*)alloc(C * C * 2);
    _Float16* WoT = (_Float16*)alloc(C * C * 2);
    _Float16* W1T = (_Float16*)alloc(F * C * 2);   // [DFF x C]
    _Float16* W2T = (_Float16*)alloc(C * F * 2);   // [C x DFF]
    _Float16* Qb  = (_Float16*)alloc(M * C * 2);
    _Float16* Kf  = (_Float16*)alloc(M * C * 2);
    _Float16* VTb = (_Float16*)alloc(M * C * 2);   // [B*H, HDIM, NTOK]
    _Float16* AO  = (_Float16*)alloc(M * C * 2);
    float*    OP  = (float*)   alloc(M * C * 4);
    float*    Hf  = (float*)   alloc(M * C * 4);
    _Float16* Hb  = (_Float16*)alloc(M * C * 2);
    _Float16* FF  = (_Float16*)alloc(M * F * 2);
    float*    F2  = (float*)   alloc(M * C * 4);

    const dim3 tb(32, 8);

    // 1) converts / weight transposes
    f32_to_f16_kernel<<<(M * C) / 1024, 256, 0, stream>>>(x, Xb, M * C);
    transpose_f16_kernel<<<dim3(C / 32, C / 32), tb, 0, stream>>>(Wq, WqT, C, C);
    transpose_f16_kernel<<<dim3(C / 32, C / 32), tb, 0, stream>>>(Wk, WkT, C, C);
    transpose_f16_kernel<<<dim3(C / 32, C / 32), tb, 0, stream>>>(Wv, WvT, C, C);
    transpose_f16_kernel<<<dim3(C / 32, C / 32), tb, 0, stream>>>(Wo, WoT, C, C);
    transpose_f16_kernel<<<dim3(F / 32, C / 32), tb, 0, stream>>>(W1, W1T, C, F);
    transpose_f16_kernel<<<dim3(C / 32, F / 32), tb, 0, stream>>>(W2, W2T, F, C);

    // 2) QKV projections (WMMA + TDM-staged B)
    gemm_f16_kernel<1><<<dim3(M / 128, C / 64), 128, 0, stream>>>(Xb, WqT, bq, Qb, M, C, C);
    gemm_f16_kernel<1><<<dim3(M / 128, C / 64), 128, 0, stream>>>(Xb, WkT, bk, Kf, M, C, C);
    gemm_f16_kernel<2><<<dim3(M / 128, C / 64), 128, 0, stream>>>(Xb, WvT, bv, VTb, M, C, C);

    // 3) flash attention (WMMA)
    attn_kernel<<<dim3(BATCH * HEADS, NTOK / 64), 128, 0, stream>>>(Qb, Kf, VTb, AO);

    // 4) output projection + residual LN1
    gemm_f16_kernel<0><<<dim3(M / 128, C / 64), 128, 0, stream>>>(AO, WoT, bo, OP, M, C, C);
    add_ln_kernel<<<M, 256, 0, stream>>>(x, OP, g1, b1, Hf, Hb);

    // 5) FFN (WMMA) + residual LN2 -> d_out
    gemm_f16_kernel<3><<<dim3(M / 128, F / 64), 128, 0, stream>>>(Hb, W1T, bf1, FF, M, F, C);
    gemm_f16_kernel<0><<<dim3(M / 128, C / 64), 128, 0, stream>>>(FF, W2T, bf2, F2, M, C, F);
    add_ln_kernel<<<M, 256, 0, stream>>>(Hf, F2, g2, b2, (float*)d_out, nullptr);
}
